// MegaCartTensorOut_54443005444464
// MI455X (gfx1250) — compile-verified, matrix operand order checked by reading, not run on hardware
//
#include <hip/hip_runtime.h>
#include <math.h>

typedef __attribute__((ext_vector_type(16))) __bf16          v16bf;
typedef __attribute__((ext_vector_type(16))) unsigned short  v16us;
typedef __attribute__((ext_vector_type(8)))  unsigned int    v8u;
typedef __attribute__((ext_vector_type(8)))  float           v8f;

#define WPB 8            // waves per block (256 threads) in the GEMM kernel
#define SPH_STRIDE 480
#define SC_STRIDE  128

#define SCHED_FENCE() __builtin_amdgcn_sched_barrier(0)

// ---------------------------------------------------------------- helpers

__device__ __forceinline__ unsigned short f2bf(float f) {   // RNE scalar
  unsigned int u = __float_as_uint(f);
  unsigned int r = u + 0x7fffu + ((u >> 16) & 1u);
  return (unsigned short)(r >> 16);
}

// two f32 -> packed bf16 (round via +0x8000 bias, then byte-perm)
__device__ __forceinline__ unsigned int pkbf(float lo, float hi) {
  unsigned int ul = __float_as_uint(lo) + 0x8000u;
  unsigned int uh = __float_as_uint(hi) + 0x8000u;
  return __builtin_amdgcn_perm(uh, ul, 0x07060302u);  // {uh[3],uh[2],ul[3],ul[2]}
}

// exact packed-bf16 -> f32 expansions (1 VALU op each)
__device__ __forceinline__ float bf_lo(unsigned int p) { return __uint_as_float(p << 16); }
__device__ __forceinline__ float bf_hi(unsigned int p) { return __uint_as_float(p & 0xffff0000u); }

__device__ __forceinline__ v16bf load_bfrag(const unsigned short* __restrict__ wreg,
                                            int tile, int lane) {
  const v16us* p = (const v16us*)(wreg + tile * 512 + lane * 16);
  return __builtin_bit_cast(v16bf, *p);
}

__device__ __forceinline__ v8f wmma_bf16(v8u a, v16bf b, v8f c) {
  return __builtin_amdgcn_wmma_f32_16x16x32_bf16(
      false, __builtin_bit_cast(v16bf, a), false, b, (short)0, c, false, false);
}

__device__ __forceinline__ float half_reduce(float v) {   // 16 lanes of a half-wave
  v += __shfl_xor(v, 1, 32);
  v += __shfl_xor(v, 2, 32);
  v += __shfl_xor(v, 4, 32);
  v += __shfl_xor(v, 8, 32);
  return v;
}

__device__ __forceinline__ float wave_reduce(float v) {   // all 32 lanes
  v += __shfl_xor(v, 1, 32);
  v += __shfl_xor(v, 2, 32);
  v += __shfl_xor(v, 4, 32);
  v += __shfl_xor(v, 8, 32);
  v += __shfl_xor(v, 16, 32);
  return v;
}

// A-fragment K mapping (CDNA5 16-bit A layout): element i -> k_local:
//   half 0 holds K 0..7 (i=0..7) and 16..23 (i=8..15); half 1: 8..15 / 24..31

__device__ __forceinline__ v8u make_afrag_cont(const float* nb, int kt, int hf) {
  const float4* pa = (const float4*)(nb + kt * 32 + hf * 8);
  const float4* pb = (const float4*)(nb + kt * 32 + 16 + hf * 8);
  float4 q0 = pa[0], q1 = pa[1], q2 = pb[0], q3 = pb[1];
  v8u a;
  a[0] = pkbf(q0.x, q0.y); a[1] = pkbf(q0.z, q0.w);
  a[2] = pkbf(q1.x, q1.y); a[3] = pkbf(q1.z, q1.w);
  a[4] = pkbf(q2.x, q2.y); a[5] = pkbf(q2.z, q2.w);
  a[6] = pkbf(q3.x, q3.y); a[7] = pkbf(q3.z, q3.w);
  return a;
}

__device__ __forceinline__ v8u make_afrag_str(const float* __restrict__ nb, int mstride,
                                              int m, int kt, int hf) {
  v8u a;
#pragma unroll
  for (int j = 0; j < 8; ++j) {
    int i0 = 2 * j, i1 = 2 * j + 1;
    int kl0 = (i0 < 8) ? (i0 + hf * 8) : (i0 + 8 + hf * 8);
    int kl1 = (i1 < 8) ? (i1 + hf * 8) : (i1 + 8 + hf * 8);
    float lo = nb[(kt * 32 + kl0) * mstride + m];
    float hi = nb[(kt * 32 + kl1) * mstride + m];
    a[j] = pkbf(lo, hi);
  }
  return a;
}

// ---------------------------------------------------------------- CG init (device-computed Wigner 3j)

__device__ double dfact(int n) { double r = 1.0; for (int i = 2; i <= n; ++i) r *= (double)i; return r; }

__device__ void build_u(int l, double Ur[5][5], double Ui[5][5]) {
  for (int i = 0; i < 5; ++i) for (int j = 0; j < 5; ++j) { Ur[i][j] = 0.0; Ui[i][j] = 0.0; }
  const double is2 = 0.70710678118654752440;
  Ur[l][l] = 1.0;
  for (int m = 1; m <= l; ++m) {
    double s = (m & 1) ? -1.0 : 1.0;
    Ur[l + m][l + m] = s * is2;
    Ur[l + m][l - m] = is2;
    Ui[l - m][l + m] = -s * is2;
    Ui[l - m][l - m] = is2;
  }
}

__device__ void compute_real_cg(int l1, int l2, int l3, double scale, float* __restrict__ out) {
  double C[5][5][5];
  for (int a = 0; a < 5; ++a) for (int b = 0; b < 5; ++b) for (int c = 0; c < 5; ++c) C[a][b][c] = 0.0;
  for (int m1 = -l1; m1 <= l1; ++m1) {
    for (int m2 = -l2; m2 <= l2; ++m2) {
      int m3 = m1 + m2;
      if (m3 < -l3 || m3 > l3) continue;
      double pre = sqrt((double)(2 * l3 + 1) * dfact(l1 + l2 - l3) * dfact(l1 - l2 + l3) *
                        dfact(-l1 + l2 + l3) / dfact(l1 + l2 + l3 + 1));
      pre *= sqrt(dfact(l3 + m3) * dfact(l3 - m3) * dfact(l1 - m1) * dfact(l1 + m1) *
                  dfact(l2 - m2) * dfact(l2 + m2));
      double s = 0.0;
      for (int k = 0; k <= l1 + l2 - l3; ++k) {
        int d0 = k, d1 = l1 + l2 - l3 - k, d2 = l1 - m1 - k, d3 = l2 + m2 - k,
            d4 = l3 - l2 + m1 + k, d5 = l3 - l1 - m2 + k;
        if (d0 < 0 || d1 < 0 || d2 < 0 || d3 < 0 || d4 < 0 || d5 < 0) continue;
        double denom = dfact(d0) * dfact(d1) * dfact(d2) * dfact(d3) * dfact(d4) * dfact(d5);
        s += ((k & 1) ? -1.0 : 1.0) / denom;
      }
      C[m1 + l1][m2 + l2][m3 + l3] = pre * s;
    }
  }
  double U1r[5][5], U1i[5][5], U2r[5][5], U2i[5][5], U3r[5][5], U3i[5][5];
  build_u(l1, U1r, U1i); build_u(l2, U2r, U2i); build_u(l3, U3r, U3i);
  int n1 = 2 * l1 + 1, n2 = 2 * l2 + 1, n3 = 2 * l3 + 1;
  double Rr[5][5][5], Ri[5][5][5];
  double mr = 0.0, mi = 0.0;
  for (int a = 0; a < n1; ++a) for (int b = 0; b < n2; ++b) for (int c = 0; c < n3; ++c) {
    double sr = 0.0, si = 0.0;
    for (int m = 0; m < n1; ++m) for (int n = 0; n < n2; ++n) for (int o = 0; o < n3; ++o) {
      double cv = C[m][n][o];
      if (cv == 0.0) continue;
      double tr = U1r[a][m] * U2r[b][n] - U1i[a][m] * U2i[b][n];
      double ti = U1r[a][m] * U2i[b][n] + U1i[a][m] * U2r[b][n];
      double qr = tr * U3r[c][o] + ti * U3i[c][o];     // * conj(U3)
      double qi = ti * U3r[c][o] - tr * U3i[c][o];
      sr += qr * cv; si += qi * cv;
    }
    Rr[a][b][c] = sr; Ri[a][b][c] = si;
    mr = fmax(mr, fabs(sr)); mi = fmax(mi, fabs(si));
  }
  bool useR = (mr >= mi);
  double nrm = 0.0;
  for (int a = 0; a < n1; ++a) for (int b = 0; b < n2; ++b) for (int c = 0; c < n3; ++c) {
    double v = useR ? Rr[a][b][c] : Ri[a][b][c];
    nrm += v * v;
  }
  double inv = scale / sqrt(nrm);
  for (int a = 0; a < n1; ++a) for (int b = 0; b < n2; ++b) for (int c = 0; c < n3; ++c)
    out[(a * n2 + b) * n3 + c] = (float)((useR ? Rr[a][b][c] : Ri[a][b][c]) * inv);
}

// cgbuf layout (floats): RCG[i] at i*128; QB[l] at 1152 + l*48
__global__ void init_cg_kernel(float* __restrict__ cgbuf) {
  int t = threadIdx.x;
  const int LAt[9] = {0, 1, 2, 1, 2, 0, 2, 1, 2};
  const int LBt[9] = {0, 1, 2, 1, 2, 2, 0, 1, 2};
  const int LOt[9] = {0, 0, 0, 1, 1, 2, 2, 2, 2};
  if (t < 9) {
    compute_real_cg(LAt[t], LBt[t], LOt[t], 1.0, cgbuf + t * 128);
  } else if (t < 12) {
    int l = t - 9;
    compute_real_cg(1, 1, l, sqrt(2.0 * l + 1.0), cgbuf + 1152 + l * 48);
  }
}

// ---------------------------------------------------------------- weight packing

__global__ void pack_weight_kernel(const float* __restrict__ W, int K, int N,
                                   unsigned short* __restrict__ dst) {
  int total = (K / 32) * (N / 16) * 512;
  int t = blockIdx.x * 256 + threadIdx.x;
  if (t >= total) return;
  int tile = t >> 9, rem = t & 511, lane = rem >> 4, i = rem & 15;
  int ntn = N / 16;
  int kt = tile / ntn, nt = tile % ntn;
  int n = lane & 15, hf = lane >> 4;
  int kl = (i < 8) ? (i + hf * 8) : (i + 8 + hf * 8);
  int k = kt * 32 + kl;
  dst[t] = f2bf(W[k * N + nt * 16 + n]);
}

__global__ void zero_kernel(float* __restrict__ p, int n) {
  int i = blockIdx.x * 256 + threadIdx.x;
  if (i < n) p[i] = 0.0f;
}

// ---------------------------------------------------------------- kernel A: GEMMs + norm, store packed records
// Record (per (node,u) pair, pair = node*32+u):
//   x0r[pair] : 1 uint  (bf16 xs0 in lo)
//   x1r[pair] : uint2   (xs1 m0,m1 | m2,-)
//   x2r[pair] : uint4   (xs2 m0,m1 | m2,m3 | m4,- | 0)
//   wr[pair*10 + i] : bf16 tp-weight of instruction i (i=0..8)
// Out-of-range rows are redirected to dump slot pair = Ntot*32.

template <int L, int KT>
__device__ __forceinline__ void compute_xs_store(const float* __restrict__ nb, int hf, int lane,
                                                 const unsigned short* __restrict__ wreg,
                                                 float invK, int t0, int Ntot,
                                                 void* __restrict__ region) {
  constexpr int M = 2 * L + 1;
  v8f acc[M][2] = {};
#pragma unroll
  for (int m = 0; m < M; ++m) {
#pragma unroll
    for (int kt = 0; kt < KT; ++kt) {
      v8u a;
      if constexpr (M == 1) a = make_afrag_cont(nb, kt, hf);
      else                  a = make_afrag_str(nb, M, m, kt, hf);
#pragma unroll
      for (int nt = 0; nt < 2; ++nt)
        acc[m][nt] = wmma_bf16(a, load_bfrag(wreg, kt * 2 + nt, lane), acc[m][nt]);
    }
  }
  float ss[8] = {};
#pragma unroll
  for (int m = 0; m < M; ++m)
#pragma unroll
    for (int nt = 0; nt < 2; ++nt)
#pragma unroll
      for (int r = 0; r < 8; ++r) { float v = acc[m][nt][r] * invK; ss[r] += v * v; }
  float ir[8];
#pragma unroll
  for (int r = 0; r < 8; ++r) {
    float tot = half_reduce(ss[r]);
    ir[r] = rsqrtf(tot / (32.0f * M) + 1e-5f) * invK;
  }
  const int dump = Ntot * 32;
#pragma unroll
  for (int nt = 0; nt < 2; ++nt) {
#pragma unroll
    for (int r = 0; r < 8; ++r) {
      int node = t0 + r + hf * 8;
      int pair = (node < Ntot) ? (node * 32 + (lane & 15) + 16 * nt) : dump;
      if constexpr (L == 0) {
        ((unsigned int*)region)[pair] = pkbf(acc[0][nt][r] * ir[r], 0.0f);
      } else if constexpr (L == 1) {
        uint2 o;
        o.x = pkbf(acc[0][nt][r] * ir[r], acc[1][nt][r] * ir[r]);
        o.y = pkbf(acc[2][nt][r] * ir[r], 0.0f);
        ((uint2*)region)[pair] = o;
      } else {
        uint4 o;
        o.x = pkbf(acc[0][nt][r] * ir[r], acc[1][nt][r] * ir[r]);
        o.y = pkbf(acc[2][nt][r] * ir[r], acc[3][nt][r] * ir[r]);
        o.z = pkbf(acc[4][nt][r] * ir[r], 0.0f);
        o.w = 0u;
        ((uint4*)region)[pair] = o;
      }
    }
  }
}

template <int I>
__device__ __forceinline__ void store_w(const v8u& h0, const v8u& h1,
                                        const unsigned short* __restrict__ wfA2,
                                        const float* __restrict__ b2,
                                        int lane, int hf, int t0, int Ntot,
                                        unsigned short* __restrict__ wr) {
  v8f wa = {}, wb = {};
  wa = wmma_bf16(h0, load_bfrag(wfA2, 2 * I,          lane), wa);
  wa = wmma_bf16(h1, load_bfrag(wfA2, 18 + 2 * I,     lane), wa);
  wb = wmma_bf16(h0, load_bfrag(wfA2, 2 * I + 1,      lane), wb);
  wb = wmma_bf16(h1, load_bfrag(wfA2, 18 + 2 * I + 1, lane), wb);
  float bb0 = b2[32 * I + (lane & 15)], bb1 = b2[32 * I + 16 + (lane & 15)];
  const int dump = Ntot * 32;
#pragma unroll
  for (int t = 0; t < 2; ++t)
#pragma unroll
    for (int r = 0; r < 8; ++r) {
      float val = t ? (wb[r] + bb1) : (wa[r] + bb0);
      int node = t0 + r + hf * 8;
      int pair = (node < Ntot) ? (node * 32 + (lane & 15) + 16 * t) : dump;
      wr[pair * 10 + I] = f2bf(val);
    }
  SCHED_FENCE();
}

__global__ void __launch_bounds__(256) gemm_kernel(
    const float* __restrict__ xsc, const float* __restrict__ xsp, int Ntot,
    const float* __restrict__ b1, const float* __restrict__ b2,
    const unsigned short* __restrict__ wf,
    unsigned int* __restrict__ x0r, uint2* __restrict__ x1r,
    uint4* __restrict__ x2r, unsigned short* __restrict__ wr) {
  __shared__ float lds_h[WPB * 16 * 64];
  const int tid = threadIdx.x;
  const int wv = tid >> 5;
  const int lane = tid & 31;
  const int hf = lane >> 4;
  const int row = lane & 15;
  const int t0 = (blockIdx.x * WPB + wv) * 16;
  const bool wactive = t0 < Ntot;
  const int nd = t0 + row;
  const bool valid = wactive && (nd < Ntot);
  const float* nsp = xsp + (size_t)(valid ? nd : 0) * SPH_STRIDE;
  const float* nsc = xsc + (size_t)(valid ? nd : 0) * SC_STRIDE;

  const unsigned short* wfW0 = wf;
  const unsigned short* wfW1 = wf + 8 * 512;
  const unsigned short* wfW2 = wf + 12 * 512;
  const unsigned short* wfA1 = wf + 14 * 512;
  const unsigned short* wfA2 = wf + 30 * 512;

  v8u hfr0 = {}, hfr1 = {};

  if (wactive) {
    compute_xs_store<0, 4>(nsp + 0,   hf, lane, wfW0, 0.08838834764f, t0, Ntot, x0r);
    SCHED_FENCE();
    compute_xs_store<1, 2>(nsp + 128, hf, lane, wfW1, 0.125f,         t0, Ntot, x1r);
    SCHED_FENCE();
    compute_xs_store<2, 1>(nsp + 320, hf, lane, wfW2, 0.1767766953f,  t0, Ntot, x2r);
    SCHED_FENCE();

    // MLP layer 1: h = silu(x_scalar @ A1 + b1)
    v8f hacc[4] = {};
#pragma unroll
    for (int kt = 0; kt < 4; ++kt) {
      v8u a = make_afrag_cont(nsc, kt, hf);
#pragma unroll
      for (int nt = 0; nt < 4; ++nt)
        hacc[nt] = wmma_bf16(a, load_bfrag(wfA1, kt * 4 + nt, lane), hacc[nt]);
    }
#pragma unroll
    for (int nt = 0; nt < 4; ++nt) {
      float bb = b1[row + 16 * nt];
#pragma unroll
      for (int r = 0; r < 8; ++r) {
        float v = hacc[nt][r] + bb;
        float h = v / (1.0f + __expf(-v));
        lds_h[wv * 1024 + (r + hf * 8) * 64 + row + 16 * nt] = h;
      }
    }
    SCHED_FENCE();
  }
  __syncthreads();   // C-layout -> A-fragment transpose through LDS
  if (wactive) {
    hfr0 = make_afrag_cont(&lds_h[wv * 1024 + row * 64], 0, hf);
    hfr1 = make_afrag_cont(&lds_h[wv * 1024 + row * 64], 1, hf);
    SCHED_FENCE();

    store_w<0>(hfr0, hfr1, wfA2, b2, lane, hf, t0, Ntot, wr);
    store_w<1>(hfr0, hfr1, wfA2, b2, lane, hf, t0, Ntot, wr);
    store_w<2>(hfr0, hfr1, wfA2, b2, lane, hf, t0, Ntot, wr);
    store_w<3>(hfr0, hfr1, wfA2, b2, lane, hf, t0, Ntot, wr);
    store_w<4>(hfr0, hfr1, wfA2, b2, lane, hf, t0, Ntot, wr);
    store_w<5>(hfr0, hfr1, wfA2, b2, lane, hf, t0, Ntot, wr);
    store_w<6>(hfr0, hfr1, wfA2, b2, lane, hf, t0, Ntot, wr);
    store_w<7>(hfr0, hfr1, wfA2, b2, lane, hf, t0, Ntot, wr);
    store_w<8>(hfr0, hfr1, wfA2, b2, lane, hf, t0, Ntot, wr);
  }
}

// ---------------------------------------------------------------- kernel B: CG contraction, one thread per (node,u)

__global__ void __launch_bounds__(256) cg_kernel(
    const unsigned int* __restrict__ x0r, const uint2* __restrict__ x1r,
    const uint4* __restrict__ x2r, const unsigned int* __restrict__ wr5,
    const int* __restrict__ batch, int Ntot,
    const float* __restrict__ cgbuf,
    const float* __restrict__ p0, const float* __restrict__ p1, const float* __restrict__ p2,
    float* __restrict__ out) {
  int t = blockIdx.x * 256 + threadIdx.x;
  int node = t >> 5;                 // wave = one node, lane = channel u
  if (node >= Ntot) return;
  int pair = t;

  float x0 = bf_lo(x0r[pair]);
  uint2 u1 = x1r[pair];
  float x1[3] = {bf_lo(u1.x), bf_hi(u1.x), bf_lo(u1.y)};
  uint4 u2 = x2r[pair];
  float x2[5] = {bf_lo(u2.x), bf_hi(u2.x), bf_lo(u2.y), bf_hi(u2.y), bf_lo(u2.z)};
  float w[9];
#pragma unroll
  for (int j = 0; j < 5; ++j) {
    unsigned int uw = wr5[pair * 5 + j];
    w[2 * j] = bf_lo(uw);
    if (2 * j + 1 < 9) w[2 * j + 1] = bf_hi(uw);
  }

  const float c0 = 0.05892556510f;   // sqrt(1/96)  * 1/sqrt(3)
  const float c1 = 0.15309310892f;   // sqrt(3)/8   * 1/sqrt(2)
  const float c2 = 0.09882117688f;   // sqrt(5/128) * 1/2
  float s0 = 0.0f, s1[3] = {}, s2[5] = {};

  { // i0: (0,0,0)
    s0 += (c0 * p0[0]) * w[0] * cgbuf[0] * x0 * x0;
  }
  { // i1: (1,1,0)
    const float* R = cgbuf + 128; float y = 0.0f;
#pragma unroll
    for (int a = 0; a < 3; ++a)
#pragma unroll
      for (int b = 0; b < 3; ++b) y += R[a * 3 + b] * x1[a] * x1[b];
    s0 += (c0 * p0[1]) * w[1] * y;
  }
  { // i2: (2,2,0)
    const float* R = cgbuf + 256; float y = 0.0f;
#pragma unroll
    for (int a = 0; a < 5; ++a)
#pragma unroll
      for (int b = 0; b < 5; ++b) y += R[a * 5 + b] * x2[a] * x2[b];
    s0 += (c0 * p0[2]) * w[2] * y;
  }
  { // i3: (1,1,1)
    const float* R = cgbuf + 384; float k = c1 * p1[0] * w[3];
#pragma unroll
    for (int c = 0; c < 3; ++c) {
      float y = 0.0f;
#pragma unroll
      for (int a = 0; a < 3; ++a)
#pragma unroll
        for (int b = 0; b < 3; ++b) y += R[(a * 3 + b) * 3 + c] * x1[a] * x1[b];
      s1[c] += k * y;
    }
  }
  { // i4: (2,2,1)
    const float* R = cgbuf + 512; float k = c1 * p1[1] * w[4];
#pragma unroll
    for (int c = 0; c < 3; ++c) {
      float y = 0.0f;
#pragma unroll
      for (int a = 0; a < 5; ++a)
#pragma unroll
        for (int b = 0; b < 5; ++b) y += R[(a * 5 + b) * 3 + c] * x2[a] * x2[b];
      s1[c] += k * y;
    }
  }
  { // i5: (0,2,2)
    const float* R = cgbuf + 640; float k = c2 * p2[0] * w[5] * x0;
#pragma unroll
    for (int c = 0; c < 5; ++c) {
      float y = 0.0f;
#pragma unroll
      for (int b = 0; b < 5; ++b) y += R[b * 5 + c] * x2[b];
      s2[c] += k * y;
    }
  }
  { // i6: (2,0,2)
    const float* R = cgbuf + 768; float k = c2 * p2[1] * w[6] * x0;
#pragma unroll
    for (int c = 0; c < 5; ++c) {
      float y = 0.0f;
#pragma unroll
      for (int a = 0; a < 5; ++a) y += R[a * 5 + c] * x2[a];
      s2[c] += k * y;
    }
  }
  { // i7: (1,1,2)
    const float* R = cgbuf + 896; float k = c2 * p2[2] * w[7];
#pragma unroll
    for (int c = 0; c < 5; ++c) {
      float y = 0.0f;
#pragma unroll
      for (int a = 0; a < 3; ++a)
#pragma unroll
        for (int b = 0; b < 3; ++b) y += R[(a * 3 + b) * 5 + c] * x1[a] * x1[b];
      s2[c] += k * y;
    }
  }
  { // i8: (2,2,2)
    const float* R = cgbuf + 1024; float k = c2 * p2[3] * w[8];
#pragma unroll
    for (int c = 0; c < 5; ++c) {
      float y = 0.0f;
#pragma unroll
      for (int a = 0; a < 5; ++a)
#pragma unroll
        for (int b = 0; b < 5; ++b) y += R[(a * 5 + b) * 5 + c] * x2[a] * x2[b];
      s2[c] += k * y;
    }
  }

  // reduce over the 32 channels of this node
  s0 = wave_reduce(s0);
#pragma unroll
  for (int c = 0; c < 3; ++c) s1[c] = wave_reduce(s1[c]);
#pragma unroll
  for (int c = 0; c < 5; ++c) s2[c] = wave_reduce(s2[c]);

  if ((t & 31) == 0) {
    const float* q0 = cgbuf + 1152;
    const float* q1 = cgbuf + 1152 + 48;
    const float* q2 = cgbuf + 1152 + 96;
    int b = batch[node];
#pragma unroll
    for (int ii = 0; ii < 3; ++ii)
#pragma unroll
      for (int jj = 0; jj < 3; ++jj) {
        int e = ii * 3 + jj;
        float acc = s0 * q0[e];
#pragma unroll
        for (int m = 0; m < 3; ++m) acc += s1[m] * q1[e * 3 + m];
#pragma unroll
        for (int m = 0; m < 5; ++m) acc += s2[m] * q2[e * 5 + m];
        atomicAdd(&out[b * 9 + ((ii + 1) % 3) * 3 + ((jj + 1) % 3)], acc);
      }
  }
}

// ---------------------------------------------------------------- launch

extern "C" void kernel_launch(void* const* d_in, const int* in_sizes, int n_in,
                              void* d_out, int out_size, void* d_ws, size_t ws_size,
                              hipStream_t stream) {
  (void)n_in; (void)ws_size;
  const float* xsc = (const float*)d_in[0];
  const float* xsp = (const float*)d_in[1];
  const int*  batch = (const int*)d_in[2];
  const float* W0 = (const float*)d_in[4];
  const float* W1 = (const float*)d_in[5];
  const float* W2 = (const float*)d_in[6];
  const float* A1 = (const float*)d_in[7];
  const float* b1 = (const float*)d_in[8];
  const float* A2 = (const float*)d_in[9];
  const float* b2 = (const float*)d_in[10];
  const float* p0 = (const float*)d_in[11];
  const float* p1 = (const float*)d_in[12];
  const float* p2 = (const float*)d_in[13];
  int N = in_sizes[0] / 128;

  float* cgbuf = (float*)d_ws;                                 // 8 KB
  unsigned short* wfrag = (unsigned short*)((char*)d_ws + 8192);

  // packed-record workspace (SoA), base at 128 KB; recs = N*32 pairs + dump slots
  size_t recs = (size_t)N * 32 + 8;
  char* base = (char*)d_ws + 131072;
  unsigned int*   x0r = (unsigned int*)base;                 // 4 B/pair
  uint2*          x1r = (uint2*)(base + recs * 4);           // 8 B/pair
  uint4*          x2r = (uint4*)(base + recs * 12);          // 16 B/pair
  unsigned short* wr  = (unsigned short*)(base + recs * 28); // 20 B/pair

  zero_kernel<<<(out_size + 255) / 256, 256, 0, stream>>>((float*)d_out, out_size);
  init_cg_kernel<<<1, 32, 0, stream>>>(cgbuf);

  auto grid = [](long long total) { return (int)((total + 255) / 256); };
  pack_weight_kernel<<<grid(4 * 2 * 512),  256, 0, stream>>>(W0, 128, 32,  wfrag + 0);
  pack_weight_kernel<<<grid(2 * 2 * 512),  256, 0, stream>>>(W1, 64,  32,  wfrag + 8  * 512);
  pack_weight_kernel<<<grid(1 * 2 * 512),  256, 0, stream>>>(W2, 32,  32,  wfrag + 12 * 512);
  pack_weight_kernel<<<grid(4 * 4 * 512),  256, 0, stream>>>(A1, 128, 64,  wfrag + 14 * 512);
  pack_weight_kernel<<<grid(2 * 18 * 512), 256, 0, stream>>>(A2, 64,  288, wfrag + 30 * 512);

  int blocksA = (N + 16 * WPB - 1) / (16 * WPB);
  gemm_kernel<<<blocksA, 256, 0, stream>>>(xsc, xsp, N, b1, b2, wfrag,
                                           x0r, x1r, x2r, wr);

  int blocksB = grid((long long)N * 32);
  cg_kernel<<<blocksB, 256, 0, stream>>>(x0r, x1r, x2r, (const unsigned int*)wr,
                                         batch, N, cgbuf, p0, p1, p2, (float*)d_out);
}
